// MegaNeRF_17514876634269
// MI455X (gfx1250) — matrix operand order repaired and checked by reading, not compiled, not run on hardware
//
#include <hip/hip_runtime.h>
#include <hip/hip_bf16.h>

// ---------------------------------------------------------------------------
// MegaNeRF MoE forward, routed + WMMA (gfx1250 / CDNA5, wave32)
// ---------------------------------------------------------------------------
#define N_PTS   32768
#define NEXP    8
#define HID     256
#define IN_DIM  111      // 63 posenc + 48 appearance
#define IN_PAD  128
#define SKIP_K  384      // 256 + 111 padded
#define COL_K   288      // 256 + 27 padded
#define COL_HID 128
#define NUM_APP 1000

#define TILE_M  32
#define NWAVES  8
#define AP      400      // LDS activation stride (halves); mult of 16 for 32B align
#define XP      144      // LDS x0 stride (halves)

typedef __attribute__((ext_vector_type(16))) _Float16 v16h;
typedef __attribute__((ext_vector_type(8)))  _Float16 v8h;
typedef __attribute__((ext_vector_type(8)))  float    v8f;

// ---------------- workspace layout (bytes, all 256-aligned) ----------------
#define OFF_CNT   0
#define OFF_IDX   256                                   // int  [E][N]
#define OFF_X0    (OFF_IDX  + NEXP*N_PTS*4)             // f16  [N][128]
#define OFF_DIR   (OFF_X0   + (size_t)N_PTS*128*2)      // f16  [N][32]
#define OFF_W0T   (OFF_DIR  + (size_t)N_PTS*32*2)       // f16  [E][256][128]
#define OFF_PRET  (OFF_W0T  + (size_t)NEXP*256*128*2)   // f16  [E][3][256][256]
#define OFF_SKPT  (OFF_PRET + (size_t)NEXP*3*256*256*2) // f16  [E][256][384]
#define OFF_POST  (OFF_SKPT + (size_t)NEXP*256*384*2)   // f16  [E][3][256][256]
#define OFF_WFT   (OFF_POST + (size_t)NEXP*3*256*256*2) // f16  [E][256][256]
#define OFF_C1T   (OFF_WFT  + (size_t)NEXP*256*256*2)   // f16  [E][128][288]
// total ~21.6 MB

// ---------------------------------------------------------------------------
__global__ void init_kernel(int* cnt) {
  if (threadIdx.x < NEXP) cnt[threadIdx.x] = 0;
}

// f32 [G][IN][OUT] -> f16 transposed [G][OUT][KP], zero-padded K.
// 32x32 LDS tiled transpose: coalesced reads (along OUT) and writes (along KP).
__global__ __launch_bounds__(256)
void cvt_T_kernel(const float* __restrict__ src, _Float16* __restrict__ dst,
                  int IN, int OUT, int KP) {
  __shared__ float t[32][33];
  const int g  = blockIdx.z;
  const int k0 = blockIdx.y * 32;
  const int o0 = blockIdx.x * 32;
  const int tx = threadIdx.x & 31;
  const int ty = threadIdx.x >> 5;   // 0..7
  const float* s = src + (size_t)g * IN * OUT;
#pragma unroll
  for (int i = 0; i < 4; ++i) {
    int k = k0 + ty + i * 8;
    int o = o0 + tx;
    t[ty + i * 8][tx] = (k < IN && o < OUT) ? s[(size_t)k * OUT + o] : 0.0f;
  }
  __syncthreads();
  _Float16* d = dst + (size_t)g * OUT * KP;
#pragma unroll
  for (int i = 0; i < 4; ++i) {
    int o = o0 + ty + i * 8;
    int k = k0 + tx;                 // k < KP by grid construction (KP % 32 == 0)
    if (o < OUT) d[(size_t)o * KP + k] = (_Float16)t[tx][ty + i * 8];
  }
}

// ---------------------------------------------------------------------------
// Routing + input-feature build. One thread per point. Features are assembled
// in register-resident v16h chunks and stored as 32B global_store_b128s.
__global__ __launch_bounds__(256)
void route_kernel(const float* __restrict__ points, const float* __restrict__ viewdirs,
                  const int* __restrict__ app_idx, const float* __restrict__ app_embed,
                  int* __restrict__ cnt, int* __restrict__ idxlist,
                  _Float16* __restrict__ x0buf, _Float16* __restrict__ dirbuf,
                  float* __restrict__ out_d, float* __restrict__ out_c) {
  int p = blockIdx.x * blockDim.x + threadIdx.x;
  if (p >= N_PTS) return;
  float x = points[3 * p + 0], y = points[3 * p + 1], z = points[3 * p + 2];
  bool fg = (x >= -80.f) && (x <= 80.f) && (y >= -80.f) && (y <= 80.f) &&
            (z >= -4.f) && (z <= 44.f);
  // argmin over 8 centroids: x in {-50,50}, y in {-75,-25,25,75}, z=20
  int e = 0; float best = 3.4e38f;
  float dzc = z - 20.f; float dz2 = dzc * dzc;
#pragma unroll
  for (int i = 0; i < 4; ++i) {
#pragma unroll
    for (int j = 0; j < 2; ++j) {
      float cx = j ? 50.f : -50.f;
      float cy = -75.f + 50.f * (float)i;
      float d = (x - cx) * (x - cx) + (y - cy) * (y - cy) + dz2;
      int id = i * 2 + j;
      if (d < best) { best = d; e = id; }
    }
  }
  // x0 row in registers: [x, {sin,cos}(f*x) x10 freqs, app(48), pad->128]
  v16h xv[8] = {};
#define XSET(idx, val) xv[(idx) >> 4][(idx) & 15] = (_Float16)(val)
  XSET(0, x); XSET(1, y); XSET(2, z);
#pragma unroll
  for (int i = 0; i < 10; ++i) {
    float f = exp2f((float)i * (10.0f / 9.0f));
    float s, c;
    __sincosf(f * x, &s, &c); XSET(3 + 6 * i + 0, s); XSET(3 + 6 * i + 3, c);
    __sincosf(f * y, &s, &c); XSET(3 + 6 * i + 1, s); XSET(3 + 6 * i + 4, c);
    __sincosf(f * z, &s, &c); XSET(3 + 6 * i + 2, s); XSET(3 + 6 * i + 5, c);
  }
  const float* app = app_embed + ((size_t)e * NUM_APP + app_idx[p]) * 48;
#pragma unroll
  for (int j = 0; j < 48; ++j) XSET(63 + j, app[j]);
#pragma unroll
  for (int c = 0; c < 8; ++c)
    *(v16h*)(x0buf + (size_t)p * IN_PAD + c * 16) = xv[c];
#undef XSET
  // direction encoding row: [d, {sin,cos}(f*d) x4 freqs, pad->32]
  float dx = viewdirs[3 * p + 0], dy = viewdirs[3 * p + 1], dv = viewdirs[3 * p + 2];
  v16h dvv[2] = {};
#define DSET(idx, val) dvv[(idx) >> 4][(idx) & 15] = (_Float16)(val)
  DSET(0, dx); DSET(1, dy); DSET(2, dv);
#pragma unroll
  for (int i = 0; i < 4; ++i) {
    float f = exp2f((float)i * (4.0f / 3.0f));
    float s, c;
    __sincosf(f * dx, &s, &c); DSET(3 + 6 * i + 0, s); DSET(3 + 6 * i + 3, c);
    __sincosf(f * dy, &s, &c); DSET(3 + 6 * i + 1, s); DSET(3 + 6 * i + 4, c);
    __sincosf(f * dv, &s, &c); DSET(3 + 6 * i + 2, s); DSET(3 + 6 * i + 5, c);
  }
#pragma unroll
  for (int c = 0; c < 2; ++c)
    *(v16h*)(dirbuf + (size_t)p * 32 + c * 16) = dvv[c];
#undef DSET
  // zero outputs (background stays zero; fg overwritten by mlp kernel)
  out_d[p] = 0.f;
  out_c[3 * p + 0] = 0.f; out_c[3 * p + 1] = 0.f; out_c[3 * p + 2] = 0.f;
  if (fg) {
    int slot = atomicAdd(&cnt[e], 1);
    idxlist[e * N_PTS + slot] = p;
  }
}

// ---------------------------------------------------------------------------
// Prefetch a weight matrix into cache while the previous layer computes.
// Lowers to global_prefetch_b8; fire-and-forget (no counter).
__device__ __forceinline__ void pf_weights(const _Float16* p, int halves) {
  int lines = halves >> 6;                       // one 128B line per 64 halves
  for (int i = threadIdx.x; i < lines; i += 256)
    __builtin_prefetch(p + (size_t)i * 64, 0, 1);
}

// ---------------------------------------------------------------------------
// One WMMA GEMM layer over a 32-point tile. Each wave owns full n-tiles and
// BOTH m-tiles, so every B fragment (global) feeds two WMMAs (m=0..15,16..31).
// A from LDS using the ISA's 8-element-chunk lane layout.
__device__ __forceinline__ void gemm32(const _Float16* __restrict__ src, int SP,
                                       const _Float16* __restrict__ wT, int KP,
                                       const float* __restrict__ bias,
                                       int nOut, int Ktiles,
                                       _Float16* __restrict__ dst, bool relu) {
  const int wave    = threadIdx.x >> 5;
  const int lane    = threadIdx.x & 31;
  const int lane16  = lane & 15;
  const int halfSel = lane >> 4;          // 0: lanes 0-15, 1: lanes 16-31
  const int c0      = halfSel * 8;        // A-fragment chunk offset
  const int nTiles  = nOut >> 4;
  const _Float16* a0row = src + (size_t)lane16 * SP;         // rows 0..15
  const _Float16* a1row = src + (size_t)(16 + lane16) * SP;  // rows 16..31
  for (int nt = wave; nt < nTiles; nt += NWAVES) {
    v8f acc0 = {}, acc1 = {};
    const _Float16* wrow = wT + (size_t)(nt * 16 + lane16) * KP + halfSel * 8;
    for (int k = 0; k < Ktiles; ++k) {
      v16h b;                                                 // 8-element-chunk layout
      *(v8h*)&b       = *(const v8h*)(wrow + k * 32);
      *((v8h*)&b + 1) = *(const v8h*)(wrow + k * 32 + 16);
      v16h a0, a1;
      *(v8h*)&a0       = *(const v8h*)(a0row + k * 32 + c0);
      *((v8h*)&a0 + 1) = *(const v8h*)(a0row + k * 32 + 16 + c0);
      *(v8h*)&a1       = *(const v8h*)(a1row + k * 32 + c0);
      *((v8h*)&a1 + 1) = *(const v8h*)(a1row + k * 32 + 16 + c0);
      acc0 = __builtin_amdgcn_wmma_f32_16x16x32_f16(false, a0, false, b,
                                                    (short)0, acc0, false, false);
      acc1 = __builtin_amdgcn_wmma_f32_16x16x32_f16(false, a1, false, b,
                                                    (short)0, acc1, false, false);
    }
    const int col  = nt * 16 + lane16;
    const float bv = bias[col];
    const int r0   = halfSel * 8;          // D: VGPR i -> rows i / i+8
#pragma unroll
    for (int i = 0; i < 8; ++i) {
      float v0 = acc0[i] + bv;
      float v1 = acc1[i] + bv;
      if (relu) { v0 = fmaxf(v0, 0.0f); v1 = fmaxf(v1, 0.0f); }
      dst[(size_t)(r0 + i) * AP + col]      = (_Float16)v0;
      dst[(size_t)(16 + r0 + i) * AP + col] = (_Float16)v1;
    }
  }
  __syncthreads();
}

// ---------------------------------------------------------------------------
__global__ __launch_bounds__(256)
void mlp_kernel(const int* __restrict__ cnt, const int* __restrict__ idxlist,
                const _Float16* __restrict__ x0buf, const _Float16* __restrict__ dirbuf,
                const _Float16* __restrict__ W0T, const _Float16* __restrict__ preT,
                const _Float16* __restrict__ skipT, const _Float16* __restrict__ postT,
                const _Float16* __restrict__ WfT, const _Float16* __restrict__ c1T,
                const float* __restrict__ b0, const float* __restrict__ b_pre,
                const float* __restrict__ bskip, const float* __restrict__ b_post,
                const float* __restrict__ Wd, const float* __restrict__ bd,
                const float* __restrict__ bf, const float* __restrict__ bc1,
                const float* __restrict__ Wc2, const float* __restrict__ bc2,
                float* __restrict__ out_d, float* __restrict__ out_c) {
  __shared__ __attribute__((aligned(32))) _Float16 bufA[TILE_M * AP];
  __shared__ __attribute__((aligned(32))) _Float16 bufB[TILE_M * AP];
  __shared__ __attribute__((aligned(32))) _Float16 x0s[TILE_M * XP];
  __shared__ int rowIdx[TILE_M];

  const int e      = blockIdx.y;
  const int count  = cnt[e];
  const int tstart = blockIdx.x * TILE_M;
  if (tstart >= count) return;                 // uniform early exit
  const int valid = min(TILE_M, count - tstart);
  const int tid   = threadIdx.x;

  const _Float16* w0   = W0T  + (size_t)e * 256 * 128;
  const _Float16* wpre = preT + (size_t)e * 3 * 65536;
  const _Float16* wskp = skipT + (size_t)e * 256 * 384;
  const _Float16* wpst = postT + (size_t)e * 3 * 65536;
  const _Float16* wf   = WfT  + (size_t)e * 65536;
  const _Float16* wc1  = c1T  + (size_t)e * 128 * 288;

  pf_weights(w0, 256 * 128);

  if (tid < TILE_M) {
    int r = tstart + tid;
    if (r >= count) r = count - 1;             // duplicate last row; never written
    rowIdx[tid] = idxlist[e * N_PTS + r];
  }
  __syncthreads();

  // stage x0: 32 rows x 128 halves, 16 halves/thread
  {
    int r = tid >> 3, c = (tid & 7) * 16;
    v16h v = *(const v16h*)(x0buf + (size_t)rowIdx[r] * IN_PAD + c);
    *(v16h*)(x0s + (size_t)r * XP + c) = v;
  }
  __syncthreads();

  // layer 0: 111(pad128) -> 256           (prefetch next layer's weights first)
  pf_weights(wpre, 65536);
  gemm32(x0s, XP, w0, 128, b0 + e * 256, 256, 4, bufA, true);
  pf_weights(wpre + 65536, 65536);
  gemm32(bufA, AP, wpre, 256, b_pre + (e * 3 + 0) * 256, 256, 8, bufB, true);
  pf_weights(wpre + 2 * 65536, 65536);
  gemm32(bufB, AP, wpre + 65536, 256, b_pre + (e * 3 + 1) * 256, 256, 8, bufA, true);
  pf_weights(wskp, 256 * 384);
  gemm32(bufA, AP, wpre + 2 * 65536, 256, b_pre + (e * 3 + 2) * 256, 256, 8, bufB, true);
  // skip concat: x0 (incl. zero pad) into cols 256..383 of bufB
  {
    int r = tid >> 3, c = (tid & 7) * 16;
    v16h v = *(const v16h*)(x0s + (size_t)r * XP + c);
    *(v16h*)(bufB + (size_t)r * AP + 256 + c) = v;
  }
  __syncthreads();
  // skip layer: 367(pad384) -> 256
  pf_weights(wpst, 65536);
  gemm32(bufB, AP, wskp, 384, bskip + e * 256, 256, 12, bufA, true);
  pf_weights(wpst + 65536, 65536);
  gemm32(bufA, AP, wpst, 256, b_post + (e * 3 + 0) * 256, 256, 8, bufB, true);
  pf_weights(wpst + 2 * 65536, 65536);
  gemm32(bufB, AP, wpst + 65536, 256, b_post + (e * 3 + 1) * 256, 256, 8, bufA, true);
  pf_weights(wf, 65536);
  gemm32(bufA, AP, wpst + 2 * 65536, 256, b_post + (e * 3 + 2) * 256, 256, 8, bufB, true);

  // density head: relu(x . Wd + bd), one lane per point (reads bufB)
  if (tid < TILE_M && tid < valid) {
    const float* wd = Wd + e * 256;
    float s = 0.f;
    for (int h = 0; h < 256; ++h) s += (float)bufB[(size_t)tid * AP + h] * wd[h];
    out_d[rowIdx[tid]] = fmaxf(s + bd[e], 0.f);
  }

  // feature layer (no relu): bufB -> bufA cols 0..255
  pf_weights(wc1, 128 * 288);
  gemm32(bufB, AP, wf, 256, bf + e * 256, 256, 8, bufA, false);
  // append dir encoding into bufA cols 256..287
  if (tid < 64) {
    int r = tid >> 1, c = (tid & 1) * 16;
    v16h v = *(const v16h*)(dirbuf + (size_t)rowIdx[r] * 32 + c);
    *(v16h*)(bufA + (size_t)r * AP + 256 + c) = v;
  }
  __syncthreads();
  // color hidden: 283(pad288) -> 128
  gemm32(bufA, AP, wc1, 288, bc1 + e * 128, 128, 9, bufB, true);

  // color head: sigmoid(h . Wc2 + bc2), 3 lanes per point
  if (tid < TILE_M * 3) {
    int p = tid / 3, c = tid - p * 3;
    if (p < valid) {
      const float* w = Wc2 + e * COL_HID * 3;
      float s = 0.f;
      for (int h = 0; h < COL_HID; ++h) s += (float)bufB[(size_t)p * AP + h] * w[h * 3 + c];
      s += bc2[e * 3 + c];
      out_c[rowIdx[p] * 3 + c] = 1.0f / (1.0f + __expf(-s));
    }
  }
}

// ---------------------------------------------------------------------------
extern "C" void kernel_launch(void* const* d_in, const int* in_sizes, int n_in,
                              void* d_out, int out_size, void* d_ws, size_t ws_size,
                              hipStream_t stream) {
  const float* points   = (const float*)d_in[0];
  const float* viewdirs = (const float*)d_in[1];
  const int*   app_idx  = (const int*)d_in[2];
  const float* app_emb  = (const float*)d_in[3];
  const float* W0   = (const float*)d_in[4];
  const float* b0   = (const float*)d_in[5];
  const float* Wpre = (const float*)d_in[6];
  const float* bpre = (const float*)d_in[7];
  const float* Wskp = (const float*)d_in[8];
  const float* bskp = (const float*)d_in[9];
  const float* Wpst = (const float*)d_in[10];
  const float* bpst = (const float*)d_in[11];
  const float* Wd   = (const float*)d_in[12];
  const float* bd   = (const float*)d_in[13];
  const float* Wf   = (const float*)d_in[14];
  const float* bf   = (const float*)d_in[15];
  const float* Wc1  = (const float*)d_in[16];
  const float* bc1  = (const float*)d_in[17];
  const float* Wc2  = (const float*)d_in[18];
  const float* bc2  = (const float*)d_in[19];

  char* ws = (char*)d_ws;
  int*      cnt  = (int*)(ws + OFF_CNT);
  int*      idx  = (int*)(ws + OFF_IDX);
  _Float16* x0b  = (_Float16*)(ws + OFF_X0);
  _Float16* dirb = (_Float16*)(ws + OFF_DIR);
  _Float16* W0T  = (_Float16*)(ws + OFF_W0T);
  _Float16* preT = (_Float16*)(ws + OFF_PRET);
  _Float16* skpT = (_Float16*)(ws + OFF_SKPT);
  _Float16* pstT = (_Float16*)(ws + OFF_POST);
  _Float16* WfT  = (_Float16*)(ws + OFF_WFT);
  _Float16* c1T  = (_Float16*)(ws + OFF_C1T);

  float* out_d = (float*)d_out;
  float* out_c = (float*)d_out + N_PTS;

  init_kernel<<<1, 32, 0, stream>>>(cnt);

  auto cvt = [&](const float* src, _Float16* dst, int G, int IN, int OUT, int KP) {
    dim3 grid(OUT / 32 + (OUT % 32 ? 1 : 0), KP / 32, G);   // KP % 32 == 0 always
    cvt_T_kernel<<<grid, 256, 0, stream>>>(src, dst, IN, OUT, KP);
  };
  cvt(W0,   W0T,  NEXP,     IN_DIM, HID,     IN_PAD);
  cvt(Wpre, preT, NEXP * 3, HID,    HID,     HID);
  cvt(Wskp, skpT, NEXP,     367,    HID,     SKIP_K);
  cvt(Wpst, pstT, NEXP * 3, HID,    HID,     HID);
  cvt(Wf,   WfT,  NEXP,     HID,    HID,     HID);
  cvt(Wc1,  c1T,  NEXP,     283,    COL_HID, COL_K);

  route_kernel<<<(N_PTS + 255) / 256, 256, 0, stream>>>(
      points, viewdirs, app_idx, app_emb, cnt, idx, x0b, dirb, out_d, out_c);

  dim3 grid((N_PTS + TILE_M - 1) / TILE_M, NEXP);
  mlp_kernel<<<grid, 256, 0, stream>>>(
      cnt, idx, x0b, dirb, W0T, preT, skpT, pstT, WfT, c1T,
      b0, bpre, bskp, bpst, Wd, bd, bf, bc1, Wc2, bc2, out_d, out_c);
}
